// RNN_60370060312995
// MI455X (gfx1250) — matrix operand order repaired and checked by reading, not compiled, not run on hardware
//
#include <hip/hip_runtime.h>
#include <math.h>

// CDNA5 / gfx1250: wave32, WMMA bf16 16x16x32, f32 accum.
typedef __attribute__((ext_vector_type(16))) __bf16 v16bf;
typedef __attribute__((ext_vector_type(8)))  float  v8f;
typedef __attribute__((ext_vector_type(4)))  unsigned int v4u;

#define B_  64
#define T_  512
#define H_  2048
#define L_  4
#define O_  1024
#define KT_ (H_ / 32)   // 64 K-chunks of 32
#define MT_ (B_ / 16)   // 4 row tiles
#define NTH_ (H_ / 16)  // 128 col tiles (hidden)
#define NTO_ (O_ / 16)  // 64 col tiles (fc)
#define GRID_SCAN 64    // persistent-scan workgroups (512 waves = 512 tiles)
#define BH_  ((size_t)B_ * H_)
#define LBH_ ((size_t)L_ * B_ * H_)

static __device__ __forceinline__ unsigned short f2bf(float f) {
    // round-to-nearest-even fp32 -> bf16
    unsigned int u = __float_as_uint(f);
    unsigned int r = (u + 0x7fffu + ((u >> 16) & 1u)) >> 16;
    return (unsigned short)r;
}
static __device__ __forceinline__ unsigned int f2bf2(float lo, float hi) {
    return (unsigned int)f2bf(lo) | ((unsigned int)f2bf(hi) << 16);
}

// ---------------------------------------------------------------------------
// Swizzle a weight matrix W[N][K] (row-major f32, used as B = W^T) into
// per-(nt,kt) 32x16 bf16 blocks in the WMMA B-fragment lane layout:
// lane L holds 16 bf16 = W[nt*16 + (L&15)][kt*32 + (L>>4)*8 + 0..7], then [kt*32 + 16 + (L>>4)*8 + 0..7]
// ---------------------------------------------------------------------------
__global__ void swizzle_B(const float* __restrict__ W, unsigned short* __restrict__ out,
                          int N, int K) {
    int tid  = blockIdx.x * blockDim.x + threadIdx.x;
    int lane = tid & 31;
    int blk  = tid >> 5;
    int ktc  = K >> 5;
    if (blk >= (N >> 4) * ktc) return;
    int kt = blk % ktc;
    int nt = blk / ktc;
    int n  = nt * 16 + (lane & 15);
    int k0 = kt * 32 + (lane >> 4) * 8;
    const float* src = W + (size_t)n * K + k0;
    unsigned short* dst = out + ((size_t)(nt * ktc + kt)) * 512 + lane * 16;
    v4u lo, hi;
#pragma unroll
    for (int j = 0; j < 4; ++j) lo[j] = f2bf2(src[2 * j], src[2 * j + 1]);
#pragma unroll
    for (int j = 0; j < 4; ++j) hi[j] = f2bf2(src[16 + 2 * j], src[17 + 2 * j]);
    *(v4u*)(dst)     = lo;
    *(v4u*)(dst + 8) = hi;
}

// ---------------------------------------------------------------------------
// Swizzle an activation matrix A[M][K] (row-major f32) into per-(mt,kt)
// 16x32 bf16 blocks in the WMMA A-fragment lane layout.
// ---------------------------------------------------------------------------
__global__ void swizzle_A(const float* __restrict__ A, unsigned short* __restrict__ out,
                          int M, int K) {
    int tid  = blockIdx.x * blockDim.x + threadIdx.x;
    int lane = tid & 31;
    int blk  = tid >> 5;
    int ktc  = K >> 5;
    if (blk >= (M >> 4) * ktc) return;
    int kt = blk % ktc;
    int mt = blk / ktc;
    int m  = mt * 16 + (lane & 15);
    int hi = lane >> 4;
    const float* src = A + (size_t)m * K + kt * 32 + hi * 8;
    unsigned short* dst = out + ((size_t)(mt * ktc + kt)) * 512 + lane * 16;
    v4u g0, g1;
#pragma unroll
    for (int j = 0; j < 4; ++j) g0[j] = f2bf2(src[2 * j], src[2 * j + 1]);
#pragma unroll
    for (int j = 0; j < 4; ++j) g1[j] = f2bf2(src[16 + 2 * j], src[17 + 2 * j]);
    *(v4u*)(dst)     = g0;
    *(v4u*)(dst + 8) = g1;
}

// ---------------------------------------------------------------------------
// Swizzle x[B,T,H] (f32): per-t A-fragment layout of the [B,H] slice.
// ---------------------------------------------------------------------------
__global__ void swizzle_x(const float* __restrict__ x, unsigned short* __restrict__ xb) {
    int tid  = blockIdx.x * blockDim.x + threadIdx.x;
    int lane = tid & 31;
    int blk  = tid >> 5;
    int kt   = blk & (KT_ - 1);
    int rem  = blk >> 6;                 // KT_ = 64
    int mt   = rem & (MT_ - 1);
    int t    = rem >> 2;                 // MT_ = 4
    int m    = mt * 16 + (lane & 15);
    int hi   = lane >> 4;
    const float* src = x + ((size_t)m * T_ + t) * H_ + kt * 32 + hi * 8;
    unsigned short* dst = xb + (size_t)t * BH_ +
                          ((size_t)(mt * KT_ + kt)) * 512 + lane * 16;
    v4u g0, g1;
#pragma unroll
    for (int j = 0; j < 4; ++j) g0[j] = f2bf2(src[2 * j], src[2 * j + 1]);
#pragma unroll
    for (int j = 0; j < 4; ++j) g1[j] = f2bf2(src[16 + 2 * j], src[17 + 2 * j]);
    *(v4u*)(dst)     = g0;
    *(v4u*)(dst + 8) = g1;
}

__global__ void init_barrier(unsigned int* cnt) {
    if (blockIdx.x == 0 && threadIdx.x == 0) *cnt = 0u;
}

// ---------------------------------------------------------------------------
// Persistent scan kernel: the whole T x L recurrence in ONE launch.
// Exactly GRID_SCAN=64 blocks x 256 threads -> 512 waves = 512 output tiles.
// All activation buffers are addressed from ONE base pointer (AB) with
// integer-offset selects, so every access stays in the global address space
// (global_load_b128, not flat_load_b128 -> no DScnt coupling in the hot loop).
//   AB layout: [0, T*B*H)            swizzled x_t slices
//              [T*B*H, +L*B*H)       hidden ping
//              [T*B*H+L*B*H, +L*B*H) hidden pong
// ---------------------------------------------------------------------------
__global__ __launch_bounds__(256)
void rnn_scan(unsigned short* __restrict__ AB,
              const unsigned short* __restrict__ WxS,
              const unsigned short* __restrict__ WhS,
              const float* __restrict__ bx,
              const float* __restrict__ bh,
              unsigned int* __restrict__ barCnt) {
    __shared__ float tile[8][16 * 17];   // per-wave 16x16 tile, padded rows
    int lane = threadIdx.x & 31;
    int wv   = threadIdx.x >> 5;
    int w    = blockIdx.x * 8 + wv;      // 0..511
    int mt   = w >> 7;                   // 0..3
    int nt   = w & 127;                  // 0..127
    int col  = lane & 15;
    int hi   = lane >> 4;

    const size_t H0OFF = (size_t)T_ * BH_;           // hidden ping offset
    const size_t aTile = ((size_t)mt * KT_) * 512 + lane * 16;
    const size_t oTile = ((size_t)(mt * KT_ + (nt >> 1))) * 512 +
                         lane * 16 + (nt & 1) * 8;
    unsigned int barTarget = 0;

    for (int t = 0; t < T_; ++t) {
        size_t pvOff = H0OFF + (size_t)(t & 1) * LBH_;
        size_t cuOff = H0OFF + (size_t)((t + 1) & 1) * LBH_;
        for (int l = 0; l < L_; ++l) {
            size_t ainOff = (l == 0) ? (size_t)t * BH_
                                     : cuOff + (size_t)(l - 1) * BH_;
            const unsigned short* aIn = AB + ainOff + aTile;
            const unsigned short* aRe = AB + pvOff + (size_t)l * BH_ + aTile;
            const unsigned short* bXw = WxS + (size_t)l * H_ * H_ +
                                        ((size_t)nt * KT_) * 512 + lane * 16;
            const unsigned short* bHw = WhS + (size_t)l * H_ * H_ +
                                        ((size_t)nt * KT_) * 512 + lane * 16;

            v8f acc = {};
            for (int kt = 0; kt < KT_; ++kt) {
                __builtin_prefetch(bXw + (kt + 8) * 512, 0, 3);
                __builtin_prefetch(bHw + (kt + 8) * 512, 0, 3);
                v16bf a0 = *(const v16bf*)(aIn + (size_t)kt * 512);
                v16bf b0 = *(const v16bf*)(bXw + (size_t)kt * 512);
                v16bf a1 = *(const v16bf*)(aRe + (size_t)kt * 512);
                v16bf b1 = *(const v16bf*)(bHw + (size_t)kt * 512);
                acc = __builtin_amdgcn_wmma_f32_16x16x32_bf16(false, a0, false, b0,
                                                              (short)0, acc, false, false);
                acc = __builtin_amdgcn_wmma_f32_16x16x32_bf16(false, a1, false, b1,
                                                              (short)0, acc, false, false);
            }

            // bias + tanh, staged through per-wave LDS tile to re-emit in the
            // A-fragment layout for the next consumer GEMM.
            int n = nt * 16 + col;
            float bias = bx[(size_t)l * H_ + n] + bh[(size_t)l * H_ + n];
            float* Tt = &tile[wv][0];
#pragma unroll
            for (int r = 0; r < 8; ++r) {
                Tt[(r + hi * 8) * 17 + col] = tanhf(acc[r] + bias);
            }
            __syncthreads();

            // Dest lane owns row m'=lane&15, k-slots (lane>>4)*8..+7 of half nt&1.
            unsigned short* dst = AB + cuOff + (size_t)l * BH_ + oTile;
            const float* srcRow = &Tt[(lane & 15) * 17 + hi * 8];
            v4u pk;
#pragma unroll
            for (int j = 0; j < 4; ++j)
                pk[j] = f2bf2(srcRow[2 * j], srcRow[2 * j + 1]);
            *(v4u*)dst = pk;   // 16B store, 16B-aligned (lane*32 + half*16)

            // ---- grid barrier (release / arrive / acquire) ----
            barTarget += GRID_SCAN;
            __threadfence();             // release: stores visible at device scope
            __syncthreads();
            if (threadIdx.x == 0) {
                atomicAdd(barCnt, 1u);
                while (__hip_atomic_load(barCnt, __ATOMIC_RELAXED,
                                         __HIP_MEMORY_SCOPE_AGENT) < barTarget) {
                    __builtin_amdgcn_s_sleep(2);
                }
            }
            __syncthreads();
            __threadfence();             // acquire: invalidate near caches
        }
    }
}

// ---------------------------------------------------------------------------
// Final FC: out[64,1024] = H3 @ fc_w^T + fc_b   (f32 output)
// ---------------------------------------------------------------------------
__global__ __launch_bounds__(256)
void rnn_fc(const unsigned short* __restrict__ A,
            const unsigned short* __restrict__ FcS,
            const float* __restrict__ fcb,
            float* __restrict__ out) {
    int lane = threadIdx.x & 31;
    int wv   = threadIdx.x >> 5;
    int w    = blockIdx.x * 8 + wv;      // 0..255
    int mt   = w >> 6;                   // 0..3
    int nt   = w & 63;                   // 0..63

    const unsigned short* aP = A   + ((size_t)mt * KT_) * 512 + lane * 16;
    const unsigned short* bP = FcS + ((size_t)nt * KT_) * 512 + lane * 16;

    v8f acc = {};
    for (int kt = 0; kt < KT_; ++kt) {
        v16bf a = *(const v16bf*)(aP + (size_t)kt * 512);
        v16bf b = *(const v16bf*)(bP + (size_t)kt * 512);
        acc = __builtin_amdgcn_wmma_f32_16x16x32_bf16(false, a, false, b,
                                                      (short)0, acc, false, false);
    }
    int col = lane & 15;
    int hi  = lane >> 4;
    int n   = nt * 16 + col;
    float bias = fcb[n];
#pragma unroll
    for (int r = 0; r < 8; ++r) {
        int m = mt * 16 + hi * 8 + r;
        out[(size_t)m * O_ + n] = acc[r] + bias;
    }
}

// ---------------------------------------------------------------------------
extern "C" void kernel_launch(void* const* d_in, const int* in_sizes, int n_in,
                              void* d_out, int out_size, void* d_ws, size_t ws_size,
                              hipStream_t stream) {
    const float* x   = (const float*)d_in[0];
    const float* h0  = (const float*)d_in[1];
    const float* Wx  = (const float*)d_in[2];
    const float* bx  = (const float*)d_in[3];
    const float* Wh  = (const float*)d_in[4];
    const float* bh  = (const float*)d_in[5];
    const float* fcw = (const float*)d_in[6];
    const float* fcb = (const float*)d_in[7];
    float* out = (float*)d_out;

    // Workspace layout (bf16 elements)
    unsigned short* WxS = (unsigned short*)d_ws;                 // L*H*H
    unsigned short* WhS = WxS + (size_t)L_ * H_ * H_;            // L*H*H
    unsigned short* FcS = WhS + (size_t)L_ * H_ * H_;            // O*H
    unsigned short* Xb  = FcS + (size_t)O_ * H_;                 // T*B*H  (AB base)
    unsigned short* Hb0 = Xb  + (size_t)T_ * BH_;                // L*B*H (ping)
    unsigned short* Hb1 = Hb0 + LBH_;                            // L*B*H (pong)
    unsigned int*  barCnt = (unsigned int*)(Hb1 + LBH_);

    // 1) One-time bf16 swizzles: weights become L2-resident WMMA-B streams.
    int gB = (NTH_ * KT_ * 32) / 256;        // 1024 blocks for HxH
    for (int l = 0; l < L_; ++l) {
        swizzle_B<<<gB, 256, 0, stream>>>(Wx + (size_t)l * H_ * H_,
                                          WxS + (size_t)l * H_ * H_, H_, H_);
        swizzle_B<<<gB, 256, 0, stream>>>(Wh + (size_t)l * H_ * H_,
                                          WhS + (size_t)l * H_ * H_, H_, H_);
    }
    swizzle_B<<<(NTO_ * KT_ * 32) / 256, 256, 0, stream>>>(fcw, FcS, O_, H_);

    // 2) T-parallel conversion: all x_t slices -> A-layout bf16.
    swizzle_x<<<((size_t)T_ * MT_ * KT_ * 32) / 256, 256, 0, stream>>>(x, Xb);

    // 3) Initial hidden state into ping buffer; zero the grid-barrier counter.
    for (int l = 0; l < L_; ++l)
        swizzle_A<<<(MT_ * KT_ * 32) / 256, 256, 0, stream>>>(
            h0 + (size_t)l * BH_, Hb0 + (size_t)l * BH_, B_, H_);
    init_barrier<<<1, 64, 0, stream>>>(barCnt);

    // 4) Entire sequential scan in ONE persistent kernel (grid-wide barriers).
    rnn_scan<<<GRID_SCAN, 256, 0, stream>>>(Xb, WxS, WhS, bx, bh, barCnt);

    // 5) Final projection: after t=T-1 (odd), last "cur" buffer is Hb0.
    rnn_fc<<<32, 256, 0, stream>>>(Hb0 + (size_t)(L_ - 1) * BH_, FcS, fcb, out);
}
